// AgnisV5_61143154426336
// MI455X (gfx1250) — matrix operand order repaired and checked by reading, not compiled, not run on hardware
//
#include <hip/hip_runtime.h>
#include <math.h>

#define V_SZ 50257
#define D_SZ 768
#define B_SZ 8
#define T_SZ 256
#define NWG  96          // phase-1 persistent workgroups
#define COLS 8           // D_SZ / NWG columns of R per workgroup
static_assert(NWG * COLS == D_SZ, "slice");

typedef _Float16 f16;
typedef __attribute__((ext_vector_type(8)))  _Float16 v8h;
typedef __attribute__((ext_vector_type(16))) _Float16 v16h;
typedef __attribute__((ext_vector_type(8)))  float    v8f;

// ---------------------------------------------------------------------------
// Phase 0a: init barrier counter + copy h0 into the recurrent h buffer
// ---------------------------------------------------------------------------
__global__ __launch_bounds__(256) void k_init(unsigned* cnt, float* h_buf,
                                              const float* __restrict__ h0) {
  if (threadIdx.x == 0) *cnt = 0u;
  for (int i = threadIdx.x; i < B_SZ * D_SZ; i += 256) h_buf[i] = h0[i];
}

// ---------------------------------------------------------------------------
// Phase 0b: row-wise L2 normalization of core_out (-> corehat) and of the
// gathered embedding rows (-> embn, [T][B][D] order). One block per row.
// ---------------------------------------------------------------------------
__global__ __launch_bounds__(256) void k_rownorm(const float* __restrict__ core_out,
                                                 const float* __restrict__ embedding,
                                                 const int*   __restrict__ token_ids,
                                                 float* __restrict__ corehat,
                                                 float* __restrict__ embn) {
  __shared__ float red[256];
  const int r = blockIdx.x;
  const float* src;
  float* dst;
  if (r < T_SZ * B_SZ) {                       // core_out rows (already [T][B][D])
    src = core_out + (size_t)r * D_SZ;
    dst = corehat + (size_t)r * D_SZ;
  } else {                                     // embedding rows per (t,b)
    int j = r - T_SZ * B_SZ;
    int t = j >> 3, b = j & 7;
    int tokv = token_ids[b * T_SZ + t];        // token_ids layout [B][T]
    src = embedding + (size_t)tokv * D_SZ;
    dst = embn + (size_t)j * D_SZ;
  }
  float ss = 0.f;
  for (int i = threadIdx.x; i < D_SZ; i += 256) { float v = src[i]; ss += v * v; }
  red[threadIdx.x] = ss;
  __syncthreads();
  for (int s = 128; s > 0; s >>= 1) {
    if (threadIdx.x < s) red[threadIdx.x] += red[threadIdx.x + s];
    __syncthreads();
  }
  const float scale = 1.0f / fmaxf(sqrtf(red[0]), 1e-12f);
  for (int i = threadIdx.x; i < D_SZ; i += 256) dst[i] = src[i] * scale;
}

// ---------------------------------------------------------------------------
// Phase 0c (optional, if workspace is big enough): embedding f32 -> f16 once,
// so the WMMA B-fragments are straight 16B loads with no per-tile conversion.
// ---------------------------------------------------------------------------
__global__ __launch_bounds__(256) void k_cvt_emb(const float* __restrict__ src,
                                                 f16* __restrict__ dst, int n8) {
  int i = blockIdx.x * 256 + threadIdx.x;      // group of 8 elements
  if (i >= n8) return;
  const float4* p = (const float4*)src + (size_t)i * 2;
  float4 a = p[0], b = p[1];
  v8h o;
  o[0] = (f16)a.x; o[1] = (f16)a.y; o[2] = (f16)a.z; o[3] = (f16)a.w;
  o[4] = (f16)b.x; o[5] = (f16)b.y; o[6] = (f16)b.z; o[7] = (f16)b.w;
  ((v8h*)dst)[i] = o;
}

// ---------------------------------------------------------------------------
// Device-wide spin barrier (monotonic counter; no reset -> no ABA).
// ---------------------------------------------------------------------------
__device__ __forceinline__ void grid_barrier(unsigned* cnt, unsigned& target) {
  __threadfence();                 // make this block's global stores visible
  __syncthreads();
  if (threadIdx.x == 0) {
    __hip_atomic_fetch_add(cnt, 1u, __ATOMIC_RELEASE, __HIP_MEMORY_SCOPE_AGENT);
    target += NWG;
    while (__hip_atomic_load(cnt, __ATOMIC_RELAXED, __HIP_MEMORY_SCOPE_AGENT) < target) {
      __builtin_amdgcn_s_sleep(2);
    }
    __threadfence();               // acquire: invalidate WGP caches
  }
  __syncthreads();
}

// ---------------------------------------------------------------------------
// Phase 1: persistent recurrence kernel. WG g owns R columns [g*8, g*8+8);
// the R slice (768x8 f32 = 24KB) lives in LDS for all 256 steps.
// Two grid barriers per step; fused(t-1) finalization overlaps step t.
// ---------------------------------------------------------------------------
__global__ __launch_bounds__(256, 1) void k_phase1(
    const float* __restrict__ embn, const float* __restrict__ corehat,
    const float* __restrict__ R0,
    const float* __restrict__ g1, const float* __restrict__ b1,
    const float* __restrict__ g2, const float* __restrict__ b2,
    float* h_buf, float* stats1, float* stats2, f16* fused16, unsigned* cnt) {
  __shared__ float Rs[COLS * D_SZ];    // [c][i], 24 KB
  __shared__ float hpS[B_SZ * D_SZ];   // 24 KB
  __shared__ float part[256];
  __shared__ float epsS[64], yS[64], zS[64];
  __shared__ float redA[192], redB[192];
  __shared__ float mS[8], rS[8], m2S[8], r2S[8];
  __shared__ float g1S[COLS], b1S[COLS], g2S[COLS], b2S[COLS];

  const int tid  = threadIdx.x;
  const int wg   = blockIdx.x;
  const int col0 = wg * COLS;
  unsigned bar_target = 0;

  for (int e = tid; e < COLS * D_SZ; e += 256) {
    int c = e / D_SZ, i = e - c * D_SZ;
    Rs[e] = R0[(size_t)i * D_SZ + col0 + c];      // Rs[c*D+i] = R0[i][col0+c]
  }
  if (tid < COLS) {
    g1S[tid] = g1[col0 + tid]; b1S[tid] = b1[col0 + tid];
    g2S[tid] = g2[col0 + tid]; b2S[tid] = b2[col0 + tid];
  }
  __syncthreads();

  const int b_ = tid & 7;
  const int c_ = (tid >> 3) & 7;
  const int h_ = tid >> 6;            // 0..3: K-segment for the matvec split
  const int bc = c_ * 8 + b_;

  for (int t = 0; t <= T_SZ; ++t) {
    // ---- finalize fused(t-1): LN2 using stats gathered before last barrier
    if (t > 0) {
      if (tid < 192) {
        int bb = tid & 7, ch = tid >> 3;
        float s = 0.f, ss = 0.f;
#pragma unroll
        for (int q = 0; q < 4; ++q) {
          int w = ch * 4 + q;
          s  += stats2[(w * 8 + bb) * 2];
          ss += stats2[(w * 8 + bb) * 2 + 1];
        }
        redA[tid] = s; redB[tid] = ss;
      }
      __syncthreads();
      if (tid < 8) {
        float s = 0.f, ss = 0.f;
        for (int ch = 0; ch < 24; ++ch) { s += redA[ch * 8 + tid]; ss += redB[ch * 8 + tid]; }
        float m = s * (1.0f / D_SZ);
        float var = ss * (1.0f / D_SZ) - m * m;
        m2S[tid] = m; r2S[tid] = rsqrtf(var + 1e-5f);
      }
      __syncthreads();
      if (tid < 64) {
        float fv = (zS[bc] - m2S[b_]) * r2S[b_] * g2S[c_] + b2S[c_];
        fused16[(size_t)((t - 1) * B_SZ + b_) * D_SZ + col0 + c_] = (f16)fv;
      }
      __syncthreads();
      if (t == T_SZ) break;
    }

    // ---- seg1: x_hat, Hebbian R update, temporal, y = core + 0.1*temporal
    for (int i = tid; i < B_SZ * D_SZ; i += 256) hpS[i] = h_buf[i];
    __syncthreads();

    {                                   // x_hat partials: (b, c, K-quarter)
      float acc = 0.f;
      const float* hp = &hpS[b_ * D_SZ];
      const float* rr = &Rs[c_ * D_SZ];
      const int i0 = h_ * 192;
#pragma unroll 4
      for (int i = i0; i < i0 + 192; ++i) acc += hp[i] * rr[i];
      part[tid] = acc;
    }
    __syncthreads();
    if (tid < 64) {
      float xh = part[tid] + part[tid + 64] + part[tid + 128] + part[tid + 192];
      float cv = corehat[(size_t)(t * B_SZ + b_) * D_SZ + col0 + c_];
      epsS[bc] = cv - xh;
    }
    __syncthreads();

#pragma unroll
    for (int j = 0; j < (COLS * D_SZ) / 256; ++j) {   // R_new = clip(.999R + 2.5e-4 dR)
      int e = tid + j * 256;
      int c = e / D_SZ, i = e - c * D_SZ;
      float dr = 0.f;
#pragma unroll
      for (int bb = 0; bb < 8; ++bb) dr += hpS[bb * D_SZ + i] * epsS[c * 8 + bb];
      float rv = 0.999f * Rs[e] + 2.5e-4f * dr;       // ETA*SURPRISE/B = .002/8
      Rs[e] = fminf(fmaxf(rv, -3.0f), 3.0f);
    }
    __syncthreads();

    {                                   // temporal = hp @ R_new
      float acc = 0.f;
      const float* hp = &hpS[b_ * D_SZ];
      const float* rr = &Rs[c_ * D_SZ];
      const int i0 = h_ * 192;
#pragma unroll 4
      for (int i = i0; i < i0 + 192; ++i) acc += hp[i] * rr[i];
      part[tid] = acc;
    }
    __syncthreads();
    if (tid < 64) {
      float tv = part[tid] + part[tid + 64] + part[tid + 128] + part[tid + 192];
      float cv = corehat[(size_t)(t * B_SZ + b_) * D_SZ + col0 + c_];
      yS[bc] = cv + 0.1f * tv;
    }
    __syncthreads();
    if (tid < 8) {                      // LN1 partial stats for this slice
      float s = 0.f, ss = 0.f;
#pragma unroll
      for (int c = 0; c < COLS; ++c) { float y = yS[c * 8 + tid]; s += y; ss += y * y; }
      stats1[(wg * 8 + tid) * 2] = s; stats1[(wg * 8 + tid) * 2 + 1] = ss;
    }
    grid_barrier(cnt, bar_target);

    // ---- seg2: LN1 -> h_t (write to h_buf), z = h_t + emb, LN2 stats
    if (tid < 192) {
      int bb = tid & 7, ch = tid >> 3;
      float s = 0.f, ss = 0.f;
#pragma unroll
      for (int q = 0; q < 4; ++q) {
        int w = ch * 4 + q;
        s  += stats1[(w * 8 + bb) * 2];
        ss += stats1[(w * 8 + bb) * 2 + 1];
      }
      redA[tid] = s; redB[tid] = ss;
    }
    __syncthreads();
    if (tid < 8) {
      float s = 0.f, ss = 0.f;
      for (int ch = 0; ch < 24; ++ch) { s += redA[ch * 8 + tid]; ss += redB[ch * 8 + tid]; }
      float m = s * (1.0f / D_SZ);
      float var = ss * (1.0f / D_SZ) - m * m;
      mS[tid] = m; rS[tid] = rsqrtf(var + 1e-5f);
    }
    __syncthreads();
    if (tid < 64) {
      float hv = (yS[bc] - mS[b_]) * rS[b_] * g1S[c_] + b1S[c_];
      h_buf[b_ * D_SZ + col0 + c_] = hv;
      zS[bc] = hv + embn[(size_t)(t * B_SZ + b_) * D_SZ + col0 + c_];
    }
    __syncthreads();
    if (tid < 8) {
      float s = 0.f, ss = 0.f;
#pragma unroll
      for (int c = 0; c < COLS; ++c) { float z = zS[c * 8 + tid]; s += z; ss += z * z; }
      stats2[(wg * 8 + tid) * 2] = s; stats2[(wg * 8 + tid) * 2 + 1] = ss;
    }
    grid_barrier(cnt, bar_target);
  }
}

// ---------------------------------------------------------------------------
// Phase 2: logits = fused16 @ embeddingT via v_wmma_f32_16x16x32_f16.
// Block = 8 waves, tile M=32 x N=256; each wave: 16(M) x 64(N), K-loop of 32.
// ---------------------------------------------------------------------------
__device__ __forceinline__ v16h cvt16(const float* __restrict__ p) {
  float4 f0 = *(const float4*)(p);
  float4 f1 = *(const float4*)(p + 4);
  float4 f2 = *(const float4*)(p + 8);
  float4 f3 = *(const float4*)(p + 12);
  v16h r;
  r[0] = (f16)f0.x; r[1] = (f16)f0.y; r[2] = (f16)f0.z; r[3] = (f16)f0.w;
  r[4] = (f16)f1.x; r[5] = (f16)f1.y; r[6] = (f16)f1.z; r[7] = (f16)f1.w;
  r[8] = (f16)f2.x; r[9] = (f16)f2.y; r[10] = (f16)f2.z; r[11] = (f16)f2.w;
  r[12] = (f16)f3.x; r[13] = (f16)f3.y; r[14] = (f16)f3.z; r[15] = (f16)f3.w;
  return r;
}

__device__ __forceinline__ v16h join16(v8h a, v8h b) {
  return __builtin_shufflevector(a, b, 0, 1, 2, 3, 4, 5, 6, 7,
                                 8, 9, 10, 11, 12, 13, 14, 15);
}

// B-source staged as f16 (preferred): pure 16B loads, no conversion.
__global__ __launch_bounds__(256) void k_logits16(const f16* __restrict__ fused16,
                                                  const f16* __restrict__ emb16,
                                                  float* __restrict__ out) {
  const int lane = threadIdx.x & 31;
  const int wave = threadIdx.x >> 5;
  const int mi = wave & 1;
  const int ng = wave >> 1;
  const int mbase = blockIdx.y * 32 + mi * 16;
  const int nbase = blockIdx.x * 256 + ng * 64;
  const int lmod = lane & 15;
  const int lhi  = lane >> 4;

  v8f acc[4] = {v8f{}, v8f{}, v8f{}, v8f{}};
  const f16* arow = fused16 + (size_t)(mbase + lmod) * D_SZ;
  const f16* brow[4];
#pragma unroll
  for (int s = 0; s < 4; ++s) {
    int v = nbase + s * 16 + lmod;
    if (v >= V_SZ) v = V_SZ - 1;                 // clamp loads; stores masked
    brow[s] = emb16 + (size_t)v * D_SZ + lhi * 16;
  }

  for (int k = 0; k < D_SZ; k += 32) {
    const int ka = k + lhi * 16;
    v16h A = join16(*(const v8h*)(arow + ka), *(const v8h*)(arow + ka + 8));
#pragma unroll
    for (int s = 0; s < 4; ++s) {
      v16h Bf = join16(*(const v8h*)(brow[s] + k), *(const v8h*)(brow[s] + k + 8));
      acc[s] = __builtin_amdgcn_wmma_f32_16x16x32_f16(
          false, A, false, Bf, (short)0, acc[s], false, false);
    }
  }

#pragma unroll
  for (int s = 0; s < 4; ++s) {
    const int n = nbase + s * 16 + lmod;
    if (n < V_SZ) {
      float* op = out + (size_t)(mbase + lhi * 8) * V_SZ + n;
#pragma unroll
      for (int r = 0; r < 8; ++r) op[(size_t)r * V_SZ] = acc[s][r];
    }
  }
}

// Fallback: B converted from f32 on the fly (small-workspace path).
__global__ __launch_bounds__(256) void k_logits32(const f16* __restrict__ fused16,
                                                  const float* __restrict__ embedding,
                                                  float* __restrict__ out) {
  const int lane = threadIdx.x & 31;
  const int wave = threadIdx.x >> 5;
  const int mi = wave & 1;
  const int ng = wave >> 1;
  const int mbase = blockIdx.y * 32 + mi * 16;
  const int nbase = blockIdx.x * 256 + ng * 64;
  const int lmod = lane & 15;
  const int lhi  = lane >> 4;

  v8f acc[4] = {v8f{}, v8f{}, v8f{}, v8f{}};
  const f16* arow = fused16 + (size_t)(mbase + lmod) * D_SZ;
  int vcols[4];
#pragma unroll
  for (int s = 0; s < 4; ++s) {
    int v = nbase + s * 16 + lmod;
    vcols[s] = (v < V_SZ) ? v : (V_SZ - 1);
  }

  for (int k = 0; k < D_SZ; k += 32) {
    const int ka = k + lhi * 16;
    v16h A = join16(*(const v8h*)(arow + ka), *(const v8h*)(arow + ka + 8));
    const int kb = k + lhi * 16;
#pragma unroll
    for (int s = 0; s < 4; ++s) {
      v16h Bf = cvt16(embedding + (size_t)vcols[s] * D_SZ + kb);
      acc[s] = __builtin_amdgcn_wmma_f32_16x16x32_f16(
          false, A, false, Bf, (short)0, acc[s], false, false);
    }
  }

#pragma unroll
  for (int s = 0; s < 4; ++s) {
    const int n = nbase + s * 16 + lmod;
    if (n < V_SZ) {
      float* op = out + (size_t)(mbase + lhi * 8) * V_SZ + n;
#pragma unroll
      for (int r = 0; r < 8; ++r) op[(size_t)r * V_SZ] = acc[s][r];
    }
  }
}

// ---------------------------------------------------------------------------
extern "C" void kernel_launch(void* const* d_in, const int* in_sizes, int n_in,
                              void* d_out, int out_size, void* d_ws, size_t ws_size,
                              hipStream_t stream) {
  (void)in_sizes; (void)n_in; (void)out_size;
  const float* embedding = (const float*)d_in[0];
  const float* R0        = (const float*)d_in[1];
  const float* h0        = (const float*)d_in[2];
  const float* r_gamma   = (const float*)d_in[3];
  const float* r_beta    = (const float*)d_in[4];
  const float* o_gamma   = (const float*)d_in[5];
  const float* o_beta    = (const float*)d_in[6];
  const float* core_out  = (const float*)d_in[7];
  const int*   token_ids = (const int*)d_in[8];
  float* out = (float*)d_out;

  char* ws = (char*)d_ws;
  size_t off = 0;
  auto take = [&](size_t bytes) {
    size_t o = off;
    off += (bytes + 255) & ~(size_t)255;
    return o;
  };
  unsigned* cnt  = (unsigned*)(ws + take(sizeof(unsigned)));
  float* embn    = (float*)(ws + take(sizeof(float) * T_SZ * B_SZ * D_SZ));
  float* corehat = (float*)(ws + take(sizeof(float) * T_SZ * B_SZ * D_SZ));
  float* h_buf   = (float*)(ws + take(sizeof(float) * B_SZ * D_SZ));
  float* stats1  = (float*)(ws + take(sizeof(float) * NWG * 8 * 2));
  float* stats2  = (float*)(ws + take(sizeof(float) * NWG * 8 * 2));
  f16*   fused16 = (f16*)(ws + take(sizeof(f16) * T_SZ * B_SZ * D_SZ));
  f16*   emb16   = (f16*)(ws + take(sizeof(f16) * (size_t)V_SZ * D_SZ));
  const bool staged = (ws_size >= off);          // enough scratch for f16 embedding?

  k_init<<<1, 256, 0, stream>>>(cnt, h_buf, h0);
  k_rownorm<<<2 * T_SZ * B_SZ, 256, 0, stream>>>(core_out, embedding, token_ids,
                                                 corehat, embn);
  if (staged) {
    const int n8 = (V_SZ * D_SZ) / 8;            // 768 % 8 == 0
    k_cvt_emb<<<(n8 + 255) / 256, 256, 0, stream>>>(embedding, emb16, n8);
  }
  k_phase1<<<NWG, 256, 0, stream>>>(embn, corehat, R0, r_gamma, r_beta, o_gamma,
                                    o_beta, h_buf, stats1, stats2, fused16, cnt);
  dim3 g2((V_SZ + 255) / 256, (T_SZ * B_SZ) / 32);   // 197 x 64 blocks
  if (staged) {
    k_logits16<<<g2, 256, 0, stream>>>(fused16, emb16, out);
  } else {
    k_logits32<<<g2, 256, 0, stream>>>(fused16, embedding, out);
  }
}
